// SelfAttention2_27839978012945
// MI455X (gfx1250) — matrix operand order repaired and checked by reading, not compiled, not run on hardware
//
#include <hip/hip_runtime.h>
#include <stdint.h>

#define NPOS 4096
#define CHD  64
#define CIN  512

typedef __attribute__((ext_vector_type(16))) __bf16 v16bf;
typedef __attribute__((ext_vector_type(8)))  float  v8f;

struct alignas(16) U4 { unsigned x, y, z, w; };
struct alignas(16) F4 { float x, y, z, w; };
union V16U { v16bf v; U4 u[2]; };

__device__ __forceinline__ v16bf ld_v16(const __bf16* p0, const __bf16* p1) {
  V16U t;
  t.u[0] = *(const U4*)p0;
  t.u[1] = *(const U4*)p1;
  return t.v;
}

__device__ __forceinline__ v8f bwmma(v16bf a, v16bf b, v8f c) {
  return __builtin_amdgcn_wmma_f32_16x16x32_bf16(false, a, false, b, (short)0, c,
                                                 false, false);
}

__device__ __forceinline__ unsigned pk2(float a, float b) {
  union { __bf16 h[2]; unsigned u; } t;
  t.h[0] = (__bf16)a;
  t.h[1] = (__bf16)b;
  return t.u;
}

__device__ __forceinline__ void lds_fence() {
  asm volatile("s_wait_dscnt 0" ::: "memory");
}

// Async DMA: 16 bytes global -> LDS per lane, tracked by ASYNCcnt.
__device__ __forceinline__ void async_cp16(unsigned lds_off, const void* g) {
  asm volatile("global_load_async_to_lds_b128 %0, %1, off"
               :: "v"(lds_off), "v"((unsigned long long)(uintptr_t)g)
               : "memory");
}

__device__ __forceinline__ void wait_async0() {
  asm volatile("s_wait_asynccnt 0" ::: "memory");
}

// ---------------- K0: weight conversion f32 -> bf16 ----------------
__global__ __launch_bounds__(256) void k_cvt(const float* Wf, const float* Wg,
                                             const float* Wh, const float* Wv,
                                             __bf16* of, __bf16* og, __bf16* oh,
                                             __bf16* ov) {
  int t = blockIdx.x * 256 + threadIdx.x;
  if (t < 32768)       of[t]          = (__bf16)Wf[t];
  else if (t < 65536)  og[t - 32768]  = (__bf16)Wg[t - 32768];
  else if (t < 98304)  oh[t - 65536]  = (__bf16)Wh[t - 65536];
  else if (t < 131072) ov[t - 98304]  = (__bf16)Wv[t - 98304];
}

// ---------------- K1: projections ff/fg (row-major [N][64]) + fh^T [64][N] ----
__global__ __launch_bounds__(256) void k_proj(
    const float* __restrict__ x,
    const __bf16* __restrict__ Wq, const __bf16* __restrict__ Wk,
    const __bf16* __restrict__ Wo,
    const float* __restrict__ bq, const float* __restrict__ bk,
    const float* __restrict__ bo,
    __bf16* __restrict__ ff, __bf16* __restrict__ fg,
    __bf16* __restrict__ fht) {
  const int b   = blockIdx.x >> 6;
  const int n0  = (blockIdx.x & 63) << 6;
  const int tid = threadIdx.x;
  const int lane = tid & 31;
  const int wid  = tid >> 5;
  const int l15  = lane & 15;
  const int hi   = lane >> 4;

  __shared__ __bf16 xs[64 * 40];  // [pos][c], padded stride 40 (80B, 16B aligned)

  const int pt     = wid & 3;
  const int ctbase = (wid >> 2) * 6;

  v8f acc[6];
#pragma unroll
  for (int i = 0; i < 6; ++i) {
    int ct = ctbase + i;
    const float* bias = (ct < 4) ? bq : (ct < 8 ? bk : bo);
    int ch0 = ((ct & 3) << 4) + (hi << 3);
    v8f a;
#pragma unroll
    for (int g = 0; g < 8; ++g) a[g] = bias[ch0 + g];
    acc[i] = a;
  }

  for (int cc = 0; cc < 16; ++cc) {
    __syncthreads();
#pragma unroll
    for (int r = 0; r < 8; ++r) {
      int e  = tid + 256 * r;
      int cl = e >> 6;
      int nl = e & 63;
      float v = x[((size_t)b * CIN + (cc * 32 + cl)) * NPOS + n0 + nl];
      xs[nl * 40 + cl] = (__bf16)v;
    }
    __syncthreads();
    const __bf16* bp = &xs[(pt * 16 + l15) * 40 + (hi ? 16 : 0)];
    v16bf Bv = ld_v16(bp, bp + 8);
#pragma unroll
    for (int i = 0; i < 6; ++i) {
      int ct = ctbase + i;
      const __bf16* W = (ct < 4) ? Wq : (ct < 8 ? Wk : Wo);
      int ch = ((ct & 3) << 4) + l15;
      const __bf16* ap = &W[(size_t)ch * CIN + cc * 32 + (hi ? 16 : 0)];
      v16bf Av = ld_v16(ap, ap + 8);
      acc[i] = bwmma(Av, Bv, acc[i]);
    }
  }

  const int pos = n0 + pt * 16 + l15;
#pragma unroll
  for (int i = 0; i < 6; ++i) {
    int ct  = ctbase + i;
    int chb = ((ct & 3) << 4) + (hi << 3);
    if (ct < 8) {
      __bf16* outp = (ct < 4) ? ff : fg;
      U4 u;
      u.x = pk2(acc[i][0], acc[i][1]);
      u.y = pk2(acc[i][2], acc[i][3]);
      u.z = pk2(acc[i][4], acc[i][5]);
      u.w = pk2(acc[i][6], acc[i][7]);
      *(U4*)&outp[((size_t)b * NPOS + pos) * CHD + chb] = u;
    } else {
#pragma unroll
      for (int g = 0; g < 8; ++g)
        fht[((size_t)b * CHD + chb + g) * NPOS + pos] = (__bf16)acc[i][g];
    }
  }
}

// ---------------- K2: fused flash-attention + out-proj + residual -------------
// WG = 8 waves, one 16-query tile per wave, shared batch. K/V streamed in 64-key
// chunks via async global->LDS DMA, double buffered, overlapped with WMMA.
__global__ __launch_bounds__(256) void k_attn(
    const __bf16* __restrict__ ff, const __bf16* __restrict__ fg,
    const __bf16* __restrict__ fht, const __bf16* __restrict__ Wv16,
    const float* __restrict__ bv, const float* __restrict__ x,
    const float* __restrict__ sigp, float* __restrict__ out) {
  const int tid  = threadIdx.x;
  const int lane = tid & 31;
  const int wid  = tid >> 5;
  const int l15  = lane & 15;
  const int hi   = lane >> 4;
  const int qt = blockIdx.x * 8 + wid;  // 0..2047 (8 waves share one batch)
  const int b  = qt >> 8;
  const int q0 = (qt & 255) << 4;
  const float sigma = sigp[0];

  __shared__ __bf16 kvbuf[2][2][64 * 64];  // [buf][K|V][64x64 bf16] = 32 KB
  __shared__ __bf16 stage[8][16 * 64];     // per-wave D->A transpose staging
  __bf16* st = stage[wid];

  const __bf16* kgb = fg + (size_t)b * NPOS * CHD;   // K rows [N][64]
  const __bf16* vgb = fht + (size_t)b * CHD * NPOS;  // V^T rows [64][N]

  auto issue_chunk = [&](int j0c, int buf) {
#pragma unroll
    for (int i = 0; i < 2; ++i) {
      int idx = tid + 256 * i;  // 0..511, 16B each
      // K chunk: 64 key rows x 128B, contiguous 8KB
      async_cp16((unsigned)(uintptr_t)&kvbuf[buf][0][idx * 8],
                 kgb + (size_t)j0c * CHD + idx * 8);
      // V chunk: 64 ch rows x 128B slice of [64][4096]
      int row  = idx >> 3;
      int cole = (idx & 7) * 8;
      async_cp16((unsigned)(uintptr_t)&kvbuf[buf][1][row * 64 + cole],
                 vgb + (size_t)row * NPOS + j0c + cole);
    }
  };

  // Q A-operands (16 rows x ch chunks {0..31, 32..63})
  const __bf16* qbase = ff + ((size_t)b * NPOS + q0 + l15) * CHD;
  v16bf Aq0 = ld_v16(qbase + (hi ? 16 : 0), qbase + (hi ? 16 : 0) + 8);
  v16bf Aq1 = ld_v16(qbase + 32 + (hi ? 16 : 0), qbase + 32 + (hi ? 16 : 0) + 8);

  v8f O0 = {}, O1 = {}, O2 = {}, O3 = {};
  float m[8], l[8];
#pragma unroll
  for (int g = 0; g < 8; ++g) { m[g] = -1e30f; l[g] = 0.0f; }

  issue_chunk(0, 0);

  for (int c64 = 0; c64 < 64; ++c64) {       // 64 chunks of 64 keys
    const int buf = c64 & 1;
    wait_async0();
    __syncthreads();                          // chunk resident for all waves
    if (c64 + 1 < 64) issue_chunk((c64 + 1) * 64, buf ^ 1);
    const __bf16* kb = &kvbuf[buf][0][0];
    const __bf16* vb = &kvbuf[buf][1][0];

#pragma unroll
    for (int s = 0; s < 2; ++s) {            // two 32-key steps per chunk
      const int jl = s * 32;
      // ---- S = Q K^T for two 16-key tiles (from LDS) ----
      v8f S0 = {}, S1 = {};
      const __bf16* k0 = kb + (jl + l15) * 64 + (hi ? 16 : 0);
      S0 = bwmma(Aq0, ld_v16(k0, k0 + 8), S0);
      S0 = bwmma(Aq1, ld_v16(k0 + 32, k0 + 40), S0);
      const __bf16* k1 = k0 + 16 * 64;
      S1 = bwmma(Aq0, ld_v16(k1, k1 + 8), S1);
      S1 = bwmma(Aq1, ld_v16(k1 + 32, k1 + 40), S1);

      // ---- online softmax ----
      float fsc[8];
#pragma unroll
      for (int g = 0; g < 8; ++g) {
        float v = fmaxf(S0[g], S1[g]);
        v = fmaxf(v, __shfl_xor(v, 1));
        v = fmaxf(v, __shfl_xor(v, 2));
        v = fmaxf(v, __shfl_xor(v, 4));
        v = fmaxf(v, __shfl_xor(v, 8));
        float mn = fmaxf(m[g], v);
        fsc[g] = __expf(m[g] - mn);
        m[g] = mn;
      }
#pragma unroll
      for (int g = 0; g < 8; ++g) {
        float p0 = __expf(S0[g] - m[g]);
        float p1 = __expf(S1[g] - m[g]);
        int row = g + (hi << 3);
        st[row * 64 + l15]      = (__bf16)p0;   // P tile 16x32, row stride 64
        st[row * 64 + 16 + l15] = (__bf16)p1;
        float sum = p0 + p1;
        sum += __shfl_xor(sum, 1);
        sum += __shfl_xor(sum, 2);
        sum += __shfl_xor(sum, 4);
        sum += __shfl_xor(sum, 8);
        l[g] = l[g] * fsc[g] + sum;
      }
#pragma unroll
      for (int g = 0; g < 8; ++g) {
        O0[g] *= fsc[g]; O1[g] *= fsc[g]; O2[g] *= fsc[g]; O3[g] *= fsc[g];
      }
      lds_fence();
      const __bf16* pp = st + l15 * 64 + (hi ? 16 : 0);
      v16bf Ap = ld_v16(pp, pp + 8);

      // ---- O += P @ V (V^T from LDS) ----
      const __bf16* v0 = vb + l15 * 64 + jl + (hi ? 16 : 0);
      O0 = bwmma(Ap, ld_v16(v0, v0 + 8), O0);
      const __bf16* v1 = v0 + 16 * 64;
      O1 = bwmma(Ap, ld_v16(v1, v1 + 8), O1);
      const __bf16* v2 = v1 + 16 * 64;
      O2 = bwmma(Ap, ld_v16(v2, v2 + 8), O2);
      const __bf16* v3 = v2 + 16 * 64;
      O3 = bwmma(Ap, ld_v16(v3, v3 + 8), O3);
    }
    __syncthreads();  // all waves done reading buf before it is refilled
  }

  // ---- normalize, stage O as out-proj A-operand ----
#pragma unroll
  for (int g = 0; g < 8; ++g) {
    float inv = 1.0f / l[g];
    O0[g] *= inv; O1[g] *= inv; O2[g] *= inv; O3[g] *= inv;
  }
#pragma unroll
  for (int g = 0; g < 8; ++g) {
    int row = g + (hi << 3);
    st[row * 64 + l15]      = (__bf16)O0[g];
    st[row * 64 + 16 + l15] = (__bf16)O1[g];
    st[row * 64 + 32 + l15] = (__bf16)O2[g];
    st[row * 64 + 48 + l15] = (__bf16)O3[g];
  }
  lds_fence();
  const __bf16* op = st + l15 * 64 + (hi ? 16 : 0);
  v16bf Ao0 = ld_v16(op, op + 8);
  v16bf Ao1 = ld_v16(op + 32, op + 40);

  // ---- fv = O @ Wv^T + bv ; out = x + sigma * fv (fused residual) ----
  for (int ct = 0; ct < 32; ++ct) {
    int c = (ct << 4) + l15;
    const __bf16* wp = Wv16 + (size_t)c * CHD + (hi ? 16 : 0);
    v16bf Bw0 = ld_v16(wp, wp + 8);
    v16bf Bw1 = ld_v16(wp + 32, wp + 40);
    float bvv = bv[c];
    v8f D;
#pragma unroll
    for (int g = 0; g < 8; ++g) D[g] = bvv;
    D = bwmma(Ao0, Bw0, D);
    D = bwmma(Ao1, Bw1, D);
    size_t base = ((size_t)b * CIN + c) * NPOS + q0 + (hi ? 8 : 0);
    F4 x0 = *(const F4*)&x[base];
    F4 x1 = *(const F4*)&x[base + 4];
    F4 r0, r1;
    r0.x = x0.x + sigma * D[0]; r0.y = x0.y + sigma * D[1];
    r0.z = x0.z + sigma * D[2]; r0.w = x0.w + sigma * D[3];
    r1.x = x1.x + sigma * D[4]; r1.y = x1.y + sigma * D[5];
    r1.z = x1.z + sigma * D[6]; r1.w = x1.w + sigma * D[7];
    *(F4*)&out[base]     = r0;
    *(F4*)&out[base + 4] = r1;
  }
}

extern "C" void kernel_launch(void* const* d_in, const int* in_sizes, int n_in,
                              void* d_out, int out_size, void* d_ws, size_t ws_size,
                              hipStream_t stream) {
  (void)in_sizes; (void)n_in; (void)out_size; (void)ws_size;
  const float* x   = (const float*)d_in[0];
  const float* Wf  = (const float*)d_in[1];
  const float* bf_ = (const float*)d_in[2];
  const float* Wg  = (const float*)d_in[3];
  const float* bg  = (const float*)d_in[4];
  const float* Wh  = (const float*)d_in[5];
  const float* bh  = (const float*)d_in[6];
  const float* Wv  = (const float*)d_in[7];
  const float* bv  = (const float*)d_in[8];
  const float* sg  = (const float*)d_in[9];
  float* out = (float*)d_out;

  char* ws = (char*)d_ws;
  __bf16* Wf16 = (__bf16*)(ws + 0);
  __bf16* Wg16 = (__bf16*)(ws + 65536);
  __bf16* Wh16 = (__bf16*)(ws + 131072);
  __bf16* Wv16 = (__bf16*)(ws + 196608);
  __bf16* ffp  = (__bf16*)(ws + 262144);                 // [8][4096][64] bf16
  __bf16* fgp  = (__bf16*)(ws + 262144 + 4194304);       // [8][4096][64] bf16
  __bf16* fhtp = (__bf16*)(ws + 262144 + 2 * 4194304);   // [8][64][4096] bf16

  k_cvt<<<512, 256, 0, stream>>>(Wf, Wg, Wh, Wv, Wf16, Wg16, Wh16, Wv16);
  k_proj<<<512, 256, 0, stream>>>(x, Wf16, Wg16, Wh16, bf_, bg, bh, ffp, fgp, fhtp);
  k_attn<<<256, 256, 0, stream>>>(ffp, fgp, fhtp, Wv16, bv, x, sg, out);
}
